// RecurrentEncoderDecoder_71631464563276
// MI455X (gfx1250) — compile-verified, matrix operand order checked by reading, not run on hardware
//
#include <hip/hip_runtime.h>

// ---------------- problem constants ----------------
constexpr int Bc   = 4;
constexpr int Nc   = 2048;
constexpr int Fc   = 64;
constexpr int Hh   = 4;     // heads
constexpr int Cc   = 32;    // channels per head
constexpr int HC   = 128;   // Hh*Cc
constexpr int XD   = 256;   // x' dim (2*HC)
constexpr int HIDc = 128;
constexpr int KG   = 384;   // XD + HIDc

// ---------------- vector types ----------------
typedef __attribute__((ext_vector_type(16))) _Float16 v16h_t;
typedef __attribute__((ext_vector_type(8)))  float    v8f_t;
typedef __attribute__((ext_vector_type(4)))  float    f32x4;

struct alignas(16) H8 { _Float16 h[8]; };
union UV16 { v16h_t v; H8 p[2]; _Float16 e[16]; };

__device__ __forceinline__ v8f_t zero8() {
  v8f_t z = {0.f,0.f,0.f,0.f,0.f,0.f,0.f,0.f};
  return z;
}

// ---------------- workspace layout (bytes) ----------------
constexpr size_t SZ_AT   = (size_t)Bc*Nc*Nc*4;          // a transposed
constexpr size_t SZ_XPT  = (size_t)Bc*Hh*Cc*Nc*2;       // xp^T f16 [b][h][c][n]
constexpr size_t SZ_E    = (size_t)Bc*Nc*4*4;           // es/en/Z (B,N,4) f32
constexpr size_t SZ_XPR  = (size_t)Bc*Nc*XD*4;          // x' f32
constexpr size_t SZ_HF   = (size_t)3*Bc*Nc*HIDc*2;      // h states f16 (3 decs)
constexpr size_t SZ_G    = (size_t)12*HIDc*HIDc*4;      // Gram matrices
constexpr size_t SZ_U    = (size_t)3*Bc*Nc*HIDc*2;      // U = h @ S, f16

constexpr size_t O_AT      = 0;
constexpr size_t O_XPT_IN  = O_AT + SZ_AT;
constexpr size_t O_XPT_OUT = O_XPT_IN + SZ_XPT;
constexpr size_t O_ES_IN   = O_XPT_OUT + SZ_XPT;
constexpr size_t O_EN_IN   = O_ES_IN + SZ_E;
constexpr size_t O_ES_OUT  = O_EN_IN + SZ_E;
constexpr size_t O_EN_OUT  = O_ES_OUT + SZ_E;
constexpr size_t O_Z_IN    = O_EN_OUT + SZ_E;
constexpr size_t O_Z_OUT   = O_Z_IN + SZ_E;
constexpr size_t O_XPRIME  = O_Z_OUT + SZ_E;
constexpr size_t O_HF      = O_XPRIME + SZ_XPR;
constexpr size_t O_G       = O_HF + SZ_HF;
constexpr size_t O_S       = O_G + SZ_G;
constexpr size_t O_U       = O_S + SZ_G;

constexpr size_t LOGITS = (size_t)Bc*Nc*Nc*3;
constexpr size_t BNH    = (size_t)Bc*Nc*HIDc;

constexpr float NEG_BIG = -1.0e9f;   // masked logit; exp() underflows to 0

// =======================================================================
// K0: transpose adjacency  aT[b][j][i] = a[b][i][j]
// =======================================================================
__global__ __launch_bounds__(256) void k_transpose(const float* __restrict__ a,
                                                   float* __restrict__ aT) {
  __shared__ float tile[32][33];
  const int tx = threadIdx.x, ty = threadIdx.y;
  const int bx = blockIdx.x, by = blockIdx.y, b = blockIdx.z;
#pragma unroll
  for (int k = 0; k < 4; ++k) {
    int row = ty + k * 8;
    tile[row][tx] = a[((size_t)b * Nc + by * 32 + row) * Nc + bx * 32 + tx];
  }
  __syncthreads();
#pragma unroll
  for (int k = 0; k < 4; ++k) {
    int row = ty + k * 8;
    aT[((size_t)b * Nc + bx * 32 + row) * Nc + by * 32 + tx] = tile[tx][row];
  }
}

// =======================================================================
// K1: xp = x@W (both GATs), e_self/e_neigh, write transposed f16 xp^T
// block: 256 threads handles 32 nodes
// =======================================================================
__global__ __launch_bounds__(256) void k_stage1(
    const float* __restrict__ x,
    const float* __restrict__ Win, const float* __restrict__ Wout,
    const float* __restrict__ as_in, const float* __restrict__ an_in,
    const float* __restrict__ as_out, const float* __restrict__ an_out,
    _Float16* __restrict__ xpT_in, _Float16* __restrict__ xpT_out,
    float* __restrict__ es_in, float* __restrict__ en_in,
    float* __restrict__ es_out, float* __restrict__ en_out) {
  __shared__ float xs[32 * 64];       // x tile
  __shared__ float xp[32 * 256];      // xp tile: [nn][g*128+hc]
  const int tid = threadIdx.x;
  const int n0 = blockIdx.x * 32;
  const int b = blockIdx.y;

  // load x tile (coalesced)
  for (int idx = tid; idx < 32 * 64; idx += 256)
    xs[idx] = x[((size_t)b * Nc + n0) * Fc + idx];
  __syncthreads();

  // phase 1: each thread computes one channel for 32 nodes
  {
    const int g  = tid >> 7;
    const int hc = tid & 127;
    const float* W = g ? Wout : Win;
    float acc[32];
#pragma unroll
    for (int nn = 0; nn < 32; ++nn) acc[nn] = 0.f;
#pragma unroll 4
    for (int f = 0; f < Fc; ++f) {
      float wv = W[f * HC + hc];
#pragma unroll
      for (int nn = 0; nn < 32; ++nn) acc[nn] += xs[nn * 64 + f] * wv;
    }
#pragma unroll
    for (int nn = 0; nn < 32; ++nn) xp[nn * 256 + tid] = acc[nn];
  }
  __syncthreads();

  // phase 2: attention logits es/en per (g, n, h)
  for (int task = tid; task < 512; task += 256) {
    const int sel = task & 1;
    const int h = (task >> 1) & 3;
    const int nn = (task >> 3) & 31;
    const int g = task >> 8;
    const float* att = g ? (sel ? an_out : as_out) : (sel ? an_in : as_in);
    float s = 0.f;
#pragma unroll
    for (int c = 0; c < Cc; ++c)
      s += xp[nn * 256 + g * 128 + h * 32 + c] * att[h * 32 + c];
    float* dst = g ? (sel ? en_out : es_out) : (sel ? en_in : es_in);
    dst[((size_t)b * Nc + n0 + nn) * 4 + h] = s;
  }

  // phase 3: write xp^T f16  [b][h][c][n]
  {
    const int g = tid >> 7;
    const int hc = tid & 127;
    _Float16* base = (g ? xpT_out : xpT_in) + ((size_t)b * HC + hc) * Nc + n0;
#pragma unroll
    for (int s8 = 0; s8 < 4; ++s8) {
      H8 t;
#pragma unroll
      for (int q = 0; q < 8; ++q)
        t.h[q] = (_Float16)xp[(s8 * 8 + q) * 256 + tid];
      *(H8*)(base + s8 * 8) = t;
    }
  }
}

// =======================================================================
// K2: softmax denominators Z[b][i][h] = sum_j exp(leaky(es_i+en_j) - mask)
// one wave per row; fully branchless (mask folded into exponent)
// =======================================================================
__global__ __launch_bounds__(256) void k_gat_pass1(
    const float* __restrict__ a, const float* __restrict__ es,
    const float* __restrict__ en, float* __restrict__ z) {
  const int tid = threadIdx.x, lane = tid & 31, wv = tid >> 5;
  const int i = blockIdx.x * 8 + wv;
  const int b = blockIdx.y;
  f32x4 esv = *(const f32x4*)(es + ((size_t)b * Nc + i) * 4);
  const float* arow = a + ((size_t)b * Nc + i) * Nc;
  float s[4] = {0.f, 0.f, 0.f, 0.f};
  for (int j = lane; j < Nc; j += 32) {
    float av = arow[j];
    f32x4 e4 = *(const f32x4*)(en + ((size_t)b * Nc + j) * 4);
    const bool edge = av > 0.5f;
#pragma unroll
    for (int h = 0; h < 4; ++h) {
      float f = esv[h] + e4[h];
      f = (f > 0.f) ? f : 0.2f * f;
      f = edge ? f : NEG_BIG;          // v_cndmask, then one v_exp_f32
      s[h] += __expf(f);
    }
  }
#pragma unroll
  for (int h = 0; h < 4; ++h)
    for (int off = 16; off > 0; off >>= 1) s[h] += __shfl_xor(s[h], off, 32);
  if (lane == 0) {
    f32x4 zr; zr[0] = s[0]; zr[1] = s[1]; zr[2] = s[2]; zr[3] = s[3];
    *(f32x4*)(z + ((size_t)b * Nc + i) * 4) = zr;
  }
}

// =======================================================================
// K3: GAT attention aggregation, WMMA f16.  One wave per (b, 16-row tile).
// A operand (weights) built in registers in the native wave32 layout:
//   lane: row = lane&15, half = lane>>4; element p -> K = p<8 ? 8*half+p
//   : 16+8*half+(p-8)  (two contiguous runs of 8 neighbors)
// =======================================================================
__global__ __launch_bounds__(128) void k_gat_pass2(
    const float* __restrict__ a, const float* __restrict__ es,
    const float* __restrict__ en, const float* __restrict__ z,
    const _Float16* __restrict__ xpT, const float* __restrict__ bias,
    float* __restrict__ xprime, int goff) {
  const int tid = threadIdx.x, lane = tid & 31, wv = tid >> 5;
  const int it = blockIdx.x * 4 + wv;
  const int b = blockIdx.y;
  const int col = lane & 15, kh = lane >> 4;
  const int i = it * 16 + col;

  f32x4 esv = *(const f32x4*)(es + ((size_t)b * Nc + i) * 4);
  f32x4 zv  = *(const f32x4*)(z  + ((size_t)b * Nc + i) * 4);
  float rz[4];
#pragma unroll
  for (int h = 0; h < 4; ++h) rz[h] = 1.0f / fmaxf(zv[h], 1e-30f);
  const float* arow = a + ((size_t)b * Nc + i) * Nc;

  v8f_t acc[4][2];
#pragma unroll
  for (int h = 0; h < 4; ++h)
#pragma unroll
    for (int ct = 0; ct < 2; ++ct) acc[h][ct] = zero8();

  for (int jb = 0; jb < Nc; jb += 32) {
    UV16 A[4];
#pragma unroll
    for (int r = 0; r < 2; ++r) {
      const int j0 = jb + r * 16 + kh * 8;
      f32x4 a0 = *(const f32x4*)(arow + j0);
      f32x4 a1 = *(const f32x4*)(arow + j0 + 4);
#pragma unroll
      for (int q = 0; q < 8; ++q) {
        const float aval = (q < 4) ? a0[q] : a1[q - 4];
        const bool edge = aval > 0.5f;
        f32x4 e4 = *(const f32x4*)(en + ((size_t)b * Nc + j0 + q) * 4);
#pragma unroll
        for (int h = 0; h < 4; ++h) {
          float f = esv[h] + e4[h];
          f = (f > 0.f) ? f : 0.2f * f;
          f = edge ? f : NEG_BIG;      // branchless mask -> exp underflow to 0
          A[h].e[r * 8 + q] = (_Float16)(__expf(f) * rz[h]);
        }
      }
    }
#pragma unroll
    for (int h = 0; h < 4; ++h)
#pragma unroll
      for (int ct = 0; ct < 2; ++ct) {
        const _Float16* bp =
            xpT + ((size_t)b * HC + h * Cc + ct * 16 + col) * Nc + jb + kh * 8;
        UV16 Bv;
        Bv.p[0] = *(const H8*)bp;
        Bv.p[1] = *(const H8*)(bp + 16);
        acc[h][ct] = __builtin_amdgcn_wmma_f32_16x16x32_f16(
            false, A[h].v, false, Bv.v, (short)0, acc[h][ct], false, false);
      }
  }
  // epilogue: bias + relu -> x'
#pragma unroll
  for (int h = 0; h < 4; ++h)
#pragma unroll
    for (int ct = 0; ct < 2; ++ct) {
      const int c = ct * 16 + col;
      const float bb = bias[h * 32 + c];
#pragma unroll
      for (int v = 0; v < 8; ++v) {
        const int ir = it * 16 + v + 8 * kh;
        float o = acc[h][ct][v] + bb;
        o = fmaxf(o, 0.f);
        xprime[((size_t)b * Nc + ir) * XD + goff + h * 32 + c] = o;
      }
    }
}

// =======================================================================
// K4: GRU cell (one launch per state). Block = one node, 128 threads.
// =======================================================================
__global__ __launch_bounds__(128) void k_gru(
    const float* __restrict__ xprime, const float* __restrict__ h_in,
    const float* __restrict__ Wu, const float* __restrict__ Wr,
    const float* __restrict__ Wc, const float* __restrict__ bu,
    const float* __restrict__ br, const float* __restrict__ bc,
    float* __restrict__ h_out, _Float16* __restrict__ hf16) {
  __shared__ float xh[KG];
  __shared__ float rh[HIDc];
  const int t = threadIdx.x;
  const int n = blockIdx.x, b = blockIdx.y;
  const size_t node = (size_t)b * Nc + n;
  xh[t] = xprime[node * XD + t];
  xh[t + 128] = xprime[node * XD + 128 + t];
  const float hv = h_in[node * HIDc + t];
  xh[256 + t] = hv;
  __syncthreads();
  float du = bu[n], dr = br[n];
#pragma unroll 4
  for (int k = 0; k < KG; ++k) {
    float xv = xh[k];
    du += xv * Wu[k * HIDc + t];
    dr += xv * Wr[k * HIDc + t];
  }
  const float u = 1.f / (1.f + __expf(-du));
  const float r = 1.f / (1.f + __expf(-dr));
  rh[t] = r * hv;
  __syncthreads();
  float dc = bc[n];
#pragma unroll 4
  for (int k = 0; k < XD; ++k) dc += xh[k] * Wc[k * HIDc + t];
#pragma unroll 4
  for (int k = 0; k < HIDc; ++k) dc += rh[k] * Wc[(XD + k) * HIDc + t];
  const float c = tanhf(dc);
  const float hn = u * hv + (1.f - u) * c;
  h_out[node * HIDc + t] = hn;
  hf16[node * HIDc + t] = (_Float16)hn;
}

// =======================================================================
// K5: Gram matrices G = h^T h  (per batch, per decoder)
// =======================================================================
__global__ __launch_bounds__(256) void k_gram(const float* __restrict__ hbase,
                                              float* __restrict__ G) {
  const int t = threadIdx.x;
  const int b = blockIdx.y, dec = blockIdx.z;
  const int p = blockIdx.x * 2 + (t >> 7);
  const int q = t & 127;
  const float* h = hbase + (size_t)dec * BNH + (size_t)b * Nc * HIDc;
  float s = 0.f;
#pragma unroll 4
  for (int n = 0; n < Nc; ++n) s += h[(size_t)n * HIDc + p] * h[(size_t)n * HIDc + q];
  G[((size_t)dec * Bc + b) * (HIDc * HIDc) + p * HIDc + q] = s;
}

// =======================================================================
// K6: Cholesky G = L L^T (in LDS, inv-diag stored on diagonal),
//     then S = L^{-T} R L^{-1} solved in global scratch.
// =======================================================================
__global__ __launch_bounds__(128) void k_chol(
    float* __restrict__ G, const float* __restrict__ rp0,
    const float* __restrict__ rp1, const float* __restrict__ rp2,
    float* __restrict__ S) {
  __shared__ float Gs[HIDc * HIDc];   // 64KB
  const int t = threadIdx.x;
  const int b = blockIdx.x, dec = blockIdx.y;
  const float* Rp = (dec == 0) ? rp0 : (dec == 1) ? rp1 : rp2;
  float* Gg = G + ((size_t)dec * Bc + b) * (HIDc * HIDc);
  float* Sg = S + ((size_t)dec * Bc + b) * (HIDc * HIDc);
  for (int idx = t; idx < HIDc * HIDc; idx += 128) Gs[idx] = Gg[idx];
  __syncthreads();
  // Cholesky (lower). Diagonal holds 1/L[k][k] after step k.
  for (int k = 0; k < HIDc; ++k) {
    if (t == 0) {
      float d = sqrtf(fmaxf(Gs[k * HIDc + k], 1e-20f));
      Gs[k * HIDc + k] = 1.f / d;
    }
    __syncthreads();
    if (t > k) Gs[t * HIDc + k] *= Gs[k * HIDc + k];
    __syncthreads();
    if (t > k) {
      float lik = Gs[t * HIDc + k];
      for (int j = k + 1; j <= t; ++j)
        Gs[t * HIDc + j] -= lik * Gs[j * HIDc + k];
    }
    __syncthreads();
  }
  // Row solve: T = Rp L^{-1}  (thread t = row)
  {
    const int r = t;
    for (int j = HIDc - 1; j >= 0; --j) {
      float s = Rp[r * HIDc + j];
      for (int k = j + 1; k < HIDc; ++k) s -= Sg[r * HIDc + k] * Gs[k * HIDc + j];
      Sg[r * HIDc + j] = s * Gs[j * HIDc + j];
    }
  }
  __threadfence();
  __syncthreads();
  // Column solve: L^T S = T  (thread t = column)
  {
    const int c = t;
    for (int i = HIDc - 1; i >= 0; --i) {
      float s = Sg[i * HIDc + c];
      for (int j = i + 1; j < HIDc; ++j) s -= Gs[j * HIDc + i] * Sg[j * HIDc + c];
      Sg[i * HIDc + c] = s * Gs[i * HIDc + i];
    }
  }
}

// =======================================================================
// K7: U = h @ S  (f16 output for WMMA)
// =======================================================================
__global__ __launch_bounds__(128) void k_U(const float* __restrict__ hbase,
                                           const float* __restrict__ S,
                                           _Float16* __restrict__ Uf) {
  __shared__ float hr[HIDc];
  const int t = threadIdx.x;
  const int n = blockIdx.x, b = blockIdx.y, dec = blockIdx.z;
  const float* h = hbase + (size_t)dec * BNH + ((size_t)b * Nc + n) * HIDc;
  const float* Sg = S + ((size_t)dec * Bc + b) * (HIDc * HIDc);
  hr[t] = h[t];
  __syncthreads();
  float s = 0.f;
#pragma unroll 4
  for (int k = 0; k < HIDc; ++k) s += hr[k] * Sg[k * HIDc + t];
  Uf[(((size_t)dec * Bc + b) * Nc + n) * HIDc + t] = (_Float16)s;
}

// =======================================================================
// K8: A = U h^T for all 3 decoders, fused epilogue (relu, zero-diag),
//     interleaved write into logits (B,N,N,3).  One wave per 16x16 tile.
// =======================================================================
__global__ __launch_bounds__(128) void k_decode(const _Float16* __restrict__ U,
                                                const _Float16* __restrict__ hf,
                                                float* __restrict__ out) {
  const int tid = threadIdx.x, lane = tid & 31, wv = tid >> 5;
  const int jt = blockIdx.x * 4 + wv;
  const int it = blockIdx.y;
  const int b = blockIdx.z;
  const int col = lane & 15, kh = lane >> 4;
  const int i = it * 16 + col;
  const int j = jt * 16 + col;

  v8f_t acc[3];
#pragma unroll
  for (int d = 0; d < 3; ++d) acc[d] = zero8();

#pragma unroll
  for (int dec = 0; dec < 3; ++dec) {
#pragma unroll
    for (int kt = 0; kt < 4; ++kt) {
      const _Float16* up =
          U + (((size_t)dec * Bc + b) * Nc + i) * HIDc + kt * 32 + kh * 8;
      UV16 Av;
      Av.p[0] = *(const H8*)up;
      Av.p[1] = *(const H8*)(up + 16);
      const _Float16* bp =
          hf + (((size_t)dec * Bc + b) * Nc + j) * HIDc + kt * 32 + kh * 8;
      UV16 Bv;
      Bv.p[0] = *(const H8*)bp;
      Bv.p[1] = *(const H8*)(bp + 16);
      acc[dec] = __builtin_amdgcn_wmma_f32_16x16x32_f16(
          false, Av.v, false, Bv.v, (short)0, acc[dec], false, false);
    }
  }
#pragma unroll
  for (int v = 0; v < 8; ++v) {
    const int ir = it * 16 + v + 8 * kh;
    float pv = acc[0][v];
    float mv = fmaxf(acc[1][v], 0.f);
    float sv = acc[2][v];
    if (ir == j) { mv = 0.f; sv = 0.f; }
    const size_t o = (((size_t)b * Nc + ir) * Nc + j) * 3;
    out[o + 0] = pv;
    out[o + 1] = mv;
    out[o + 2] = sv;
  }
}

// =======================================================================
extern "C" void kernel_launch(void* const* d_in, const int* in_sizes, int n_in,
                              void* d_out, int out_size, void* d_ws, size_t ws_size,
                              hipStream_t stream) {
  (void)in_sizes; (void)n_in; (void)out_size; (void)ws_size;
  const float* x    = (const float*)d_in[0];
  const float* a    = (const float*)d_in[1];
  const float* h_p  = (const float*)d_in[2];
  const float* h_mu = (const float*)d_in[3];
  const float* h_sg = (const float*)d_in[4];
  const float* gin_W  = (const float*)d_in[5];
  const float* gin_as = (const float*)d_in[6];
  const float* gin_an = (const float*)d_in[7];
  const float* gin_b  = (const float*)d_in[8];
  const float* gout_W  = (const float*)d_in[9];
  const float* gout_as = (const float*)d_in[10];
  const float* gout_an = (const float*)d_in[11];
  const float* gout_b  = (const float*)d_in[12];
  const float* p_Wu = (const float*)d_in[13];
  const float* p_bu = (const float*)d_in[14];
  const float* p_Wr = (const float*)d_in[15];
  const float* p_br = (const float*)d_in[16];
  const float* p_Wc = (const float*)d_in[17];
  const float* p_bc = (const float*)d_in[18];
  const float* m_Wu = (const float*)d_in[19];
  const float* m_bu = (const float*)d_in[20];
  const float* m_Wr = (const float*)d_in[21];
  const float* m_br = (const float*)d_in[22];
  const float* m_Wc = (const float*)d_in[23];
  const float* m_bc = (const float*)d_in[24];
  const float* s_Wu = (const float*)d_in[25];
  const float* s_bu = (const float*)d_in[26];
  const float* s_Wr = (const float*)d_in[27];
  const float* s_br = (const float*)d_in[28];
  const float* s_Wc = (const float*)d_in[29];
  const float* s_bc = (const float*)d_in[30];
  const float* R_p  = (const float*)d_in[31];
  const float* R_mu = (const float*)d_in[32];
  const float* R_sg = (const float*)d_in[33];

  char* ws = (char*)d_ws;
  float*     aT      = (float*)(ws + O_AT);
  _Float16*  xpT_in  = (_Float16*)(ws + O_XPT_IN);
  _Float16*  xpT_out = (_Float16*)(ws + O_XPT_OUT);
  float*     es_in   = (float*)(ws + O_ES_IN);
  float*     en_in   = (float*)(ws + O_EN_IN);
  float*     es_out  = (float*)(ws + O_ES_OUT);
  float*     en_out  = (float*)(ws + O_EN_OUT);
  float*     z_in    = (float*)(ws + O_Z_IN);
  float*     z_out   = (float*)(ws + O_Z_OUT);
  float*     xprime  = (float*)(ws + O_XPRIME);
  _Float16*  hf16    = (_Float16*)(ws + O_HF);
  float*     Gw      = (float*)(ws + O_G);
  float*     Sw      = (float*)(ws + O_S);
  _Float16*  Uf      = (_Float16*)(ws + O_U);

  float* out = (float*)d_out;
  float* hbase = out + LOGITS;                    // hp, hmu, hsg sections

  // 0) aT
  k_transpose<<<dim3(Nc / 32, Nc / 32, Bc), dim3(32, 8), 0, stream>>>(a, aT);
  // 1) xp / attention logits for both GATs
  k_stage1<<<dim3(Nc / 32, Bc), 256, 0, stream>>>(
      x, gin_W, gout_W, gin_as, gin_an, gout_as, gout_an,
      xpT_in, xpT_out, es_in, en_in, es_out, en_out);
  // 2) softmax denominators
  k_gat_pass1<<<dim3(Nc / 8, Bc), 256, 0, stream>>>(a, es_in, en_in, z_in);
  k_gat_pass1<<<dim3(Nc / 8, Bc), 256, 0, stream>>>(aT, es_out, en_out, z_out);
  // 3) attention aggregation (WMMA) -> x'
  k_gat_pass2<<<dim3(Nc / 64, Bc), 128, 0, stream>>>(
      a, es_in, en_in, z_in, xpT_in, gin_b, xprime, 0);
  k_gat_pass2<<<dim3(Nc / 64, Bc), 128, 0, stream>>>(
      aT, es_out, en_out, z_out, xpT_out, gout_b, xprime, HC);
  // 4) GRUs -> h states (f32 into d_out, f16 into ws)
  k_gru<<<dim3(Nc, Bc), 128, 0, stream>>>(xprime, h_p, p_Wu, p_Wr, p_Wc,
                                          p_bu, p_br, p_bc,
                                          hbase + 0 * BNH, hf16 + 0 * BNH);
  k_gru<<<dim3(Nc, Bc), 128, 0, stream>>>(xprime, h_mu, m_Wu, m_Wr, m_Wc,
                                          m_bu, m_br, m_bc,
                                          hbase + 1 * BNH, hf16 + 1 * BNH);
  k_gru<<<dim3(Nc, Bc), 128, 0, stream>>>(xprime, h_sg, s_Wu, s_Wr, s_Wc,
                                          s_bu, s_br, s_bc,
                                          hbase + 2 * BNH, hf16 + 2 * BNH);
  // 5) bilinear decoders via Gram/Cholesky:  A = h (L^-T R L^-1) h^T
  k_gram<<<dim3(HIDc / 2, Bc, 3), 256, 0, stream>>>(hbase, Gw);
  k_chol<<<dim3(Bc, 3), 128, 0, stream>>>(Gw, R_p, R_mu, R_sg, Sw);
  k_U<<<dim3(Nc, Bc, 3), 128, 0, stream>>>(hbase, Sw, Uf);
  // 6) fused U h^T for 3 decoders + epilogue -> logits
  k_decode<<<dim3(Nc / 64, Nc / 16, Bc), 128, 0, stream>>>(Uf, hf16, out);
}
